// ScaledDotProductAttention_2645699854442
// MI455X (gfx1250) — matrix operand order repaired and checked by reading, not compiled, not run on hardware
//
#include <hip/hip_runtime.h>

typedef __attribute__((ext_vector_type(16))) _Float16 v16h;
typedef __attribute__((ext_vector_type(8)))  _Float16 v8h;
typedef __attribute__((ext_vector_type(2)))  _Float16 v2h;
typedef __attribute__((ext_vector_type(2)))  __fp16   v2fp;
typedef __attribute__((ext_vector_type(8)))  float    v8f;
typedef __attribute__((ext_vector_type(4)))  float    v4f;
typedef __attribute__((ext_vector_type(4)))  int      v4i;

#define WAVES_PER_BLOCK 8

// packed f32 -> 2x f16 convert (v_cvt_pk_rtz_f16_f32), bit-cast to _Float16x2
static __device__ __forceinline__ v2h pk2h(float a, float b) {
    v2fp t = __builtin_amdgcn_cvt_pkrtz(a, b);
    return __builtin_bit_cast(v2h, t);
}

// Flash attention: B=4, H=16, S=2048, D=64, fp32 in/out.
// One block (8 wave32) owns one (b,h); each wave a 16-row query tile.
// KV consumed in 64-key tiles staged cooperatively through LDS in f16.
// Softmax runs in the A-layout domain (lane owns one query row, split with
// lane^16). Mask is applied immediately on load (s = -inf => p = 0 for free).
// The softmax denominator is accumulated by WMMA (P @ ones column).
__global__ __launch_bounds__(256) void fa_wmma_kernel(
    const float* __restrict__ Q, const float* __restrict__ K,
    const float* __restrict__ V, const int* __restrict__ MASK,
    float* __restrict__ O)
{
    constexpr int S = 2048, D = 64;

    __shared__ __align__(16) _Float16 kt[64 * 64];                  // [key][dim]
    __shared__ __align__(16) _Float16 vt[64 * 64];                  // [dim][key]
    __shared__ __align__(16) float    p_lds[WAVES_PER_BLOCK][16 * 64];

    const int tid  = threadIdx.x;
    const int lane = tid & 31;
    const int wid  = tid >> 5;
    const int h    = lane >> 4;   // half-wave id (0/1)
    const int n    = lane & 15;   // lane within half

    const int bh = blockIdx.x >> 4;          // (b*H + head); 16 blocks per bh
    const int qg = blockIdx.x & 15;
    const int b  = bh >> 4;                  // H = 16
    const int q0 = (qg * WAVES_PER_BLOCK + wid) * 16;

    // softmax scale folded into Q, in log2 domain: 1/sqrt(64) * log2(e)
    const float QSCALE = 0.125f * 1.44269504088896340736f;

    const float* qbase = Q + (size_t)bh * S * D;
    const float* kbase = K + (size_t)bh * S * D;
    const float* vbase = V + (size_t)bh * S * D;
    const int*   mbase = MASK + (size_t)b * S * S;
    float*       obase = O + (size_t)bh * S * D;

    // cooperative fill: each thread handles 16 dims of one key row
    const int kl = tid >> 2;          // key_local 0..63
    const int dq = (tid & 3) * 16;    // dim start 0,16,32,48

    // ---- Q tile (16 x 64) -> pre-scaled f16 A-layout, two K=32 chunks.
    v16h aq[2];
    {
        const float* qr = qbase + (q0 + n) * D;
        #pragma unroll
        for (int c = 0; c < 2; ++c) {
            v4f r0a = *(const v4f*)(qr + 32 * c + 16 * h);
            v4f r0b = *(const v4f*)(qr + 32 * c + 16 * h + 4);
            v4f r1a = *(const v4f*)(qr + 32 * c + 16 * h + 8);
            v4f r1b = *(const v4f*)(qr + 32 * c + 16 * h + 12);
            #pragma unroll
            for (int w = 0; w < 4; ++w) {
                aq[c][w]      = (_Float16)(r0a[w] * QSCALE);
                aq[c][4 + w]  = (_Float16)(r0b[w] * QSCALE);
                aq[c][8 + w]  = (_Float16)(r1a[w] * QSCALE);
                aq[c][12 + w] = (_Float16)(r1b[w] * QSCALE);
            }
        }
    }

    // B operand that is all-ones in column N=0: row sums of A land in D[:,0]
    v16h bv1;
    {
        const _Float16 one = (n == 0) ? (_Float16)1.0f : (_Float16)0.0f;
        #pragma unroll
        for (int i = 0; i < 16; ++i) bv1[i] = one;
    }

    // Per-lane online-softmax max for query row (q0 + n), shared with lane^16.
    // Denominator lives in acc_l (C-layout column 0, lanes 0/16).
    float m_run = -1e30f;
    v8f acc[4], acc_l = (v8f)0.0f;
    #pragma unroll
    for (int nc = 0; nc < 4; ++nc) acc[nc] = (v8f)0.0f;

    float* plds = &p_lds[wid][0];
    const int* mrow = mbase + (size_t)(q0 + n) * S;   // this lane's mask row

    for (int kv0 = 0; kv0 < S; kv0 += 64) {
        // ===== cooperative K/V tile stage (f32 -> f16, once per block) =====
        {
            const float* ks = kbase + (size_t)(kv0 + kl) * D + dq;
            const float* vs = vbase + (size_t)(kv0 + kl) * D + dq;
            v4f k0 = *(const v4f*)(ks);
            v4f k1 = *(const v4f*)(ks + 4);
            v4f k2 = *(const v4f*)(ks + 8);
            v4f k3 = *(const v4f*)(ks + 12);
            union { v8h v; v2h h2[4]; } ka, kb;
            #pragma unroll
            for (int w = 0; w < 2; ++w) {
                ka.h2[w]     = pk2h(k0[2 * w], k0[2 * w + 1]);
                ka.h2[2 + w] = pk2h(k1[2 * w], k1[2 * w + 1]);
                kb.h2[w]     = pk2h(k2[2 * w], k2[2 * w + 1]);
                kb.h2[2 + w] = pk2h(k3[2 * w], k3[2 * w + 1]);
            }
            *(v8h*)(kt + kl * 64 + dq)     = ka.v;           // row-major K
            *(v8h*)(kt + kl * 64 + dq + 8) = kb.v;

            v4f u0 = *(const v4f*)(vs);
            v4f u1 = *(const v4f*)(vs + 4);
            v4f u2 = *(const v4f*)(vs + 8);
            v4f u3 = *(const v4f*)(vs + 12);
            v2h vp[8];
            #pragma unroll
            for (int w = 0; w < 2; ++w) {
                vp[w]     = pk2h(u0[2 * w], u0[2 * w + 1]);
                vp[2 + w] = pk2h(u1[2 * w], u1[2 * w + 1]);
                vp[4 + w] = pk2h(u2[2 * w], u2[2 * w + 1]);
                vp[6 + w] = pk2h(u3[2 * w], u3[2 * w + 1]);
            }
            #pragma unroll
            for (int j = 0; j < 8; ++j) {                    // transposed V
                vt[(dq + 2 * j)     * 64 + kl] = vp[j][0];
                vt[(dq + 2 * j + 1) * 64 + kl] = vp[j][1];
            }
        }
        __syncthreads();

        // ================= Q @ K^T (scores already in log2 units) =========
        v8f sc[4];
        #pragma unroll
        for (int t = 0; t < 4; ++t) {
            v8f c8 = (v8f)0.0f;
            #pragma unroll
            for (int c = 0; c < 2; ++c) {
                v16h bk = *(const v16h*)(kt + (16 * t + n) * 64 + 32 * c + 16 * h);
                c8 = __builtin_amdgcn_wmma_f32_16x16x32_f16(
                        false, aq[c], false, bk, (short)0, c8, false, false);
            }
            sc[t] = c8;
        }

        // ---- C-layout -> A-layout: dump raw f32 scores to LDS
        #pragma unroll
        for (int t = 0; t < 4; ++t)
            #pragma unroll
            for (int r = 0; r < 8; ++r)
                plds[(8 * h + r) * 64 + 16 * t + n] = sc[t][r];

        // same-wave DS ops are in-order; wait for stores, block compiler motion
        asm volatile("s_wait_dscnt 0x0" ::: "memory");

        // ---- A-domain: lane owns row n; keys kv0 + {8h, 16+8h, 32+8h, 48+8h}+0..7
        float s[32];
        {
            const float* pr = plds + n * 64;
            #pragma unroll
            for (int g = 0; g < 4; ++g) {
                v4f a0 = *(const v4f*)(pr + 16 * g + 8 * h);
                v4f a1 = *(const v4f*)(pr + 16 * g + 8 * h + 4);
                #pragma unroll
                for (int w = 0; w < 4; ++w) {
                    s[8 * g + w]     = a0[w];
                    s[8 * g + 4 + w] = a1[w];
                }
            }
        }
        // ---- mask applied immediately (mask regs die right here);
        //      s = -inf  =>  p = exp2(-inf - m) = 0 downstream for free
        {
            const int* mp = mrow + kv0;
            #pragma unroll
            for (int g = 0; g < 4; ++g) {
                v4i m0 = *(const v4i*)(mp + 16 * g + 8 * h);
                v4i m1 = *(const v4i*)(mp + 16 * g + 8 * h + 4);
                #pragma unroll
                for (int w = 0; w < 4; ++w) {
                    s[8 * g + w]     = (m0[w] == 0) ? -INFINITY : s[8 * g + w];
                    s[8 * g + 4 + w] = (m1[w] == 0) ? -INFINITY : s[8 * g + 4 + w];
                }
            }
        }

        // ---- row max: triples fold to v_max3_f32, then one cross-half shuffle
        float gm[4];
        #pragma unroll
        for (int g = 0; g < 4; ++g) {
            const float* sg = s + 8 * g;
            float a = fmaxf(fmaxf(sg[0], sg[1]), sg[2]);
            float c = fmaxf(fmaxf(sg[3], sg[4]), sg[5]);
            float e = fmaxf(fmaxf(a, sg[6]), sg[7]);
            gm[g] = fmaxf(c, e);
        }
        float mx = fmaxf(fmaxf(fmaxf(gm[0], gm[1]), gm[2]), gm[3]);
        mx = fmaxf(mx, __shfl_xor(mx, 16, 32));
        const float m_new = fmaxf(m_run, mx);
        const float alpha = __builtin_amdgcn_exp2f(m_run - m_new);
        m_run = m_new;

        // ---- p = exp2(s - m)  (masked entries are already -inf -> 0)
        float p[32];
        #pragma unroll
        for (int i = 0; i < 32; ++i)
            p[i] = __builtin_amdgcn_exp2f(s[i] - m_new);

        // ---- pack P into two WMMA A operands (keys 0..31 and 32..63)
        union { v16h v; v2h h2[8]; } ap[2];
        #pragma unroll
        for (int k2 = 0; k2 < 2; ++k2)
            #pragma unroll
            for (int i = 0; i < 8; ++i)
                ap[k2].h2[i] = pk2h(p[16 * k2 + 2 * i], p[16 * k2 + 2 * i + 1]);

        // ---- broadcast alpha to C-layout rows, rescale acc and denominator
        float alpha_c[8];
        #pragma unroll
        for (int r = 0; r < 8; ++r) alpha_c[r] = __shfl(alpha, 8 * h + r, 32);
        #pragma unroll
        for (int nc = 0; nc < 4; ++nc)
            #pragma unroll
            for (int r = 0; r < 8; ++r) acc[nc][r] *= alpha_c[r];
        #pragma unroll
        for (int r = 0; r < 8; ++r) acc_l[r] *= alpha_c[r];

        // ================= P @ V  (+ P @ ones for the denominator) =========
        #pragma unroll
        for (int k2 = 0; k2 < 2; ++k2) {
            #pragma unroll
            for (int nc = 0; nc < 4; ++nc) {
                v8h bv0 = *(const v8h*)(vt + (16 * nc + n) * 64 + 32 * k2 + 8 * h);
                v8h bv8 = *(const v8h*)(vt + (16 * nc + n) * 64 + 32 * k2 + 16 + 8 * h);
                v16h bv = __builtin_shufflevector(bv0, bv8, 0, 1, 2, 3, 4, 5, 6, 7, 8, 9, 10, 11, 12, 13, 14, 15);
                acc[nc] = __builtin_amdgcn_wmma_f32_16x16x32_f16(
                            false, ap[k2].v, false, bv, (short)0, acc[nc], false, false);
            }
            acc_l = __builtin_amdgcn_wmma_f32_16x16x32_f16(
                        false, ap[k2].v, false, bv1, (short)0, acc_l, false, false);
        }

        __syncthreads();   // all reads done before next tile overwrites kt/vt
    }

    // ---- epilogue: O = acc / l.  l for row 8h+r sits in lane 16h, reg r.
    float rl_c[8];
    #pragma unroll
    for (int r = 0; r < 8; ++r)
        rl_c[r] = 1.0f / __shfl(acc_l[r], 16 * h, 32);
    #pragma unroll
    for (int nc = 0; nc < 4; ++nc)
        #pragma unroll
        for (int r = 0; r < 8; ++r)
            obase[(size_t)(q0 + 8 * h + r) * D + 16 * nc + n] = acc[nc][r] * rl_c[r];
}

extern "C" void kernel_launch(void* const* d_in, const int* in_sizes, int n_in,
                              void* d_out, int out_size, void* d_ws, size_t ws_size,
                              hipStream_t stream) {
    (void)in_sizes; (void)n_in; (void)out_size; (void)d_ws; (void)ws_size;
    const float* q    = (const float*)d_in[0];
    const float* k    = (const float*)d_in[1];
    const float* v    = (const float*)d_in[2];
    const int*   mask = (const int*)d_in[3];
    float*       out  = (float*)d_out;

    // B*H = 64 heads, 16 blocks per head (8 q-tiles of 16 rows per block)
    dim3 grid(64 * 16);
    dim3 block(256);
    fa_wmma_kernel<<<grid, block, 0, stream>>>(q, k, v, mask, out);
}
